// DecoderLayer_3934190043908
// MI455X (gfx1250) — matrix operand order repaired and checked by reading, not compiled, not run on hardware
//
#include <hip/hip_runtime.h>
#include <hip/hip_bf16.h>
#include <math.h>
#include <stdint.h>

// ---------------------------------------------------------------------------
// MI455X (gfx1250) transformer decoder layer.
// All GEMMs + attention use v_wmma_f32_16x16x32_bf16 (wave32).
// Straight-copy LDS staging uses GLOBAL_LOAD_ASYNC_TO_LDS_B128 (ASYNCcnt
// path) via the gfx1250 builtin; fallback is batched sync copies.
// ---------------------------------------------------------------------------

typedef __bf16 bf16;
typedef bf16  v16bf __attribute__((ext_vector_type(16)));
typedef float v8f   __attribute__((ext_vector_type(8)));

#if __has_builtin(__builtin_amdgcn_global_load_async_to_lds_b128) && \
    __has_builtin(__builtin_amdgcn_s_wait_asynccnt)
#define HAS_ASYNC_LDS 1
// Builtin signature (from clang diagnostics): param0 = int4 in AS(1) (global),
// param1 = int4 in AS(3) (LDS), then imm offset, imm cpol.
typedef int vint4 __attribute__((vector_size(4 * sizeof(int))));
typedef vint4 __attribute__((address_space(1))) vint4_g;
typedef vint4 __attribute__((address_space(3))) vint4_l;

// Per-lane 16-byte DMA copy global -> LDS (tracked by ASYNCcnt).
// AS casts go through integers: global flat == AS1 numerically; flat LDS
// addresses map to LDS via addr[31:0] (ISA 10.2 aperture rule).
static __device__ __forceinline__ void async_copy_b128(const bf16* g, bf16* l) {
    __builtin_amdgcn_global_load_async_to_lds_b128(
        (vint4_g*)(unsigned long long)(uintptr_t)g,
        (vint4_l*)(unsigned)(uintptr_t)l, 0, 0);
}
static __device__ __forceinline__ void async_wait0() {
    __builtin_amdgcn_s_wait_asynccnt(0);
}
#else
#define HAS_ASYNC_LDS 0
#endif

union FragU {
    v16bf v;
    uint4 q[2];
};

static __device__ __forceinline__ v8f zero8() {
    v8f z = {0.f, 0.f, 0.f, 0.f, 0.f, 0.f, 0.f, 0.f};
    return z;
}

static __device__ __forceinline__ v8f wmma_bf16(v16bf a, v16bf b, v8f c) {
    // 8 args: (neg_a, A, neg_b, B, c_mod, C, reuse_a, reuse_b)
    return __builtin_amdgcn_wmma_f32_16x16x32_bf16(
        false, a, false, b, (short)0, c, false, false);
}

// A-fragment (16x32 bf16, row-major source with leading dim `ld` elements).
// ISA layout: lanes 0-15 row M=lane hold K={0..7, 16..23}; lanes 16-31 hold
// K={8..15, 24..31} of row M=lane-16.  Per lane: two contiguous 16-B reads.
static __device__ __forceinline__ v16bf load_a_frag(const bf16* p, int row0,
                                                    int ld, int lane) {
    const int r   = row0 + (lane & 15);
    const int khi = (lane >> 4) & 1;
    FragU f;
    const bf16* base = p + (size_t)r * ld + khi * 8;
    f.q[0] = *(const uint4*)(base);        // K = khi*8 .. khi*8+7
    f.q[1] = *(const uint4*)(base + 16);   // K = 16+khi*8 .. +7
    return f.v;
}

// B-fragment (32x16 bf16) from a COLUMN-major staged tile: p[n*ldk + k].
// ISA layout: lanes 0-15 column N=lane hold K={0..7, 16..23}; lanes 16-31 hold
// K={8..15, 24..31} of column N=lane-16 -> per lane two contiguous 16-B reads.
static __device__ __forceinline__ v16bf load_b_fragT(const bf16* p, int col0,
                                                     int ldk, int lane) {
    const int c  = col0 + (lane & 15);
    const int kh = ((lane >> 4) & 1) * 8;
    FragU f;
    const bf16* base = p + (size_t)c * ldk + kh;
    f.q[0] = *(const uint4*)(base);
    f.q[1] = *(const uint4*)(base + 16);
    return f.v;
}

// ---------------------------------------------------------------------------
// f32 -> bf16 conversion
// ---------------------------------------------------------------------------
__global__ void f32_to_bf16_kernel(const float* __restrict__ in,
                                   bf16* __restrict__ out, int n) {
    int i = blockIdx.x * blockDim.x + threadIdx.x;
    const int stride = gridDim.x * blockDim.x;
    for (; i < n; i += stride) out[i] = (bf16)in[i];
}

// ---------------------------------------------------------------------------
// LayerNorm (torch-style: unbiased std, eps added to std) -> bf16
// One block (256 threads) per row of 1024.
// ---------------------------------------------------------------------------
__global__ __launch_bounds__(256)
void layernorm_kernel(const float* __restrict__ x, const float* __restrict__ alpha,
                      const float* __restrict__ beta, bf16* __restrict__ out) {
    __shared__ float red[16];
    const int row = blockIdx.x, tid = threadIdx.x;
    const int lane = tid & 31, wid = tid >> 5;

    const float4 v = ((const float4*)(x + (size_t)row * 1024))[tid];
    float s = v.x + v.y + v.z + v.w;
    float q = v.x * v.x + v.y * v.y + v.z * v.z + v.w * v.w;
    for (int d = 1; d < 32; d <<= 1) {
        s += __shfl_xor(s, d, 32);
        q += __shfl_xor(q, d, 32);
    }
    if (lane == 0) { red[wid] = s; red[8 + wid] = q; }
    __syncthreads();
    float S = 0.f, Q = 0.f;
#pragma unroll
    for (int i = 0; i < 8; ++i) { S += red[i]; Q += red[8 + i]; }
    const float mean = S * (1.0f / 1024.0f);
    float var = (Q - 1024.0f * mean * mean) * (1.0f / 1023.0f);
    var = fmaxf(var, 0.0f);
    const float inv = 1.0f / (sqrtf(var) + 1e-6f);

    const float4 av = ((const float4*)alpha)[tid];
    const float4 bv = ((const float4*)beta)[tid];
    bf16* o = out + (size_t)row * 1024 + tid * 4;
    o[0] = (bf16)(av.x * (v.x - mean) * inv + bv.x);
    o[1] = (bf16)(av.y * (v.y - mean) * inv + bv.y);
    o[2] = (bf16)(av.z * (v.z - mean) * inv + bv.z);
    o[3] = (bf16)(av.w * (v.w - mean) * inv + bv.w);
}

// ---------------------------------------------------------------------------
// WMMA GEMM: C[M,N] = A[M,K](bf16) * W[K,N](bf16) + bias, fused epilogue.
// Block tile 128x128, K-tile 64; 8 waves (4x2), each wave 32x64.
// 16 WMMAs per stage between barriers; next K-tile prefetched.
// MODE 0: out bf16 = v            (Q/K/V projections, FF input)
// MODE 1: out f32  = v + res      (attention out-proj + residual)
// MODE 2: out bf16 = relu(v)      (FF hidden)
// MODE 3: out f32  = res + tanh(v)(final mean head + residual)
// ---------------------------------------------------------------------------
template <int MODE>
__global__ __launch_bounds__(256)
void gemm_bf16_kernel(const bf16* __restrict__ A, const bf16* __restrict__ W,
                      const float* __restrict__ bias, const float* __restrict__ res,
                      float* __restrict__ outF, bf16* __restrict__ outB,
                      int M, int N, int K) {
    __shared__ alignas(16) bf16 As[128 * 64];   // [m][k], row-major
    __shared__ alignas(16) bf16 Bs[128 * 64];   // [n][k], transposed at staging

    const int tid  = threadIdx.x;
    const int lane = tid & 31, wid = tid >> 5;
    const int wm = wid & 3;        // 0..3 -> 32-row slab
    const int wn = wid >> 2;       // 0..1 -> 64-col slab
    const int bm = blockIdx.x * 128, bn = blockIdx.y * 128;

    v8f acc[2][4];
#pragma unroll
    for (int mi = 0; mi < 2; ++mi)
#pragma unroll
        for (int ni = 0; ni < 4; ++ni) acc[mi][ni] = zero8();

    for (int k0 = 0; k0 < K; k0 += 64) {
        __syncthreads();
        // ---- stage A: 128x64 straight copy, two b128 per thread ----
#if HAS_ASYNC_LDS
#pragma unroll
        for (int tt = 0; tt < 2; ++tt) {
            const int task = tid + tt * 256;
            const int ar = task >> 2, ac = (task & 3) * 16;
            async_copy_b128(A + (size_t)(bm + ar) * K + k0 + ac,
                            As + ar * 64 + ac);
        }
#else
        uint4 areg[2];
#pragma unroll
        for (int tt = 0; tt < 2; ++tt) {
            const int task = tid + tt * 256;
            const int ar = task >> 2, ac = (task & 3) * 16;
            areg[tt] = *(const uint4*)(A + (size_t)(bm + ar) * K + k0 + ac);
        }
#pragma unroll
        for (int tt = 0; tt < 2; ++tt) {
            const int task = tid + tt * 256;
            const int ar = task >> 2, ac = (task & 3) * 16;
            *(uint4*)(As + ar * 64 + ac) = areg[tt];
        }
#endif
        // ---- stage B transposed: batched loads, then packed b32 stores ----
        uint4 brow0[2], brow1[2];
#pragma unroll
        for (int tt = 0; tt < 2; ++tt) {
            const int task = tid + tt * 256;
            const int kr = (task >> 4) * 2;          // k row pair
            const int c0 = (task & 15) * 8;          // n within tile
            brow0[tt] = *(const uint4*)(W + (size_t)(k0 + kr) * N + bn + c0);
            brow1[tt] = *(const uint4*)(W + (size_t)(k0 + kr + 1) * N + bn + c0);
        }
#pragma unroll
        for (int tt = 0; tt < 2; ++tt) {
            const int task = tid + tt * 256;
            const int kr = (task >> 4) * 2;
            const int c0 = (task & 15) * 8;
            const unsigned short* s0 = (const unsigned short*)&brow0[tt];
            const unsigned short* s1 = (const unsigned short*)&brow1[tt];
#pragma unroll
            for (int i = 0; i < 8; ++i) {
                const unsigned pk = (unsigned)s0[i] | ((unsigned)s1[i] << 16);
                *(unsigned*)(Bs + (c0 + i) * 64 + kr) = pk;
            }
        }
        if (k0 + 64 < K) {   // prefetch next K-tile (global_prefetch_b8)
            __builtin_prefetch(A + (size_t)(bm + (tid >> 2)) * K + k0 + 64, 0, 1);
            __builtin_prefetch(W + (size_t)(k0 + 64 + (tid >> 4) * 2) * N + bn, 0, 1);
        }
#if HAS_ASYNC_LDS
        async_wait0();       // A-tile DMA complete before barrier
#endif
        __syncthreads();

#pragma unroll
        for (int s = 0; s < 2; ++s) {           // two 32-wide K slices
            v16bf af[2], bfr[4];
            af[0] = load_a_frag(As + s * 32, wm * 32, 64, lane);
            af[1] = load_a_frag(As + s * 32, wm * 32 + 16, 64, lane);
#pragma unroll
            for (int ni = 0; ni < 4; ++ni)
                bfr[ni] = load_b_fragT(Bs + s * 32, wn * 64 + ni * 16, 64, lane);
#pragma unroll
            for (int mi = 0; mi < 2; ++mi)
#pragma unroll
                for (int ni = 0; ni < 4; ++ni)
                    acc[mi][ni] = wmma_bf16(af[mi], bfr[ni], acc[mi][ni]);
        }
    }

    // Epilogue. C layout: lanes 0-15 hold rows r, lanes 16-31 rows r+8.
    const int rbase = (lane >> 4) * 8;
    const int cl = lane & 15;
#pragma unroll
    for (int mi = 0; mi < 2; ++mi) {
#pragma unroll
        for (int ni = 0; ni < 4; ++ni) {
            const int col = bn + wn * 64 + ni * 16 + cl;
            const float bv = bias[col];
#pragma unroll
            for (int r = 0; r < 8; ++r) {
                const int row = bm + wm * 32 + mi * 16 + rbase + r;
                const size_t idx = (size_t)row * N + col;
                const float v = acc[mi][ni][r] + bv;
                if constexpr (MODE == 0)      outB[idx] = (bf16)v;
                else if constexpr (MODE == 1) outF[idx] = v + res[idx];
                else if constexpr (MODE == 2) outB[idx] = (bf16)fmaxf(v, 0.0f);
                else                          outF[idx] = res[idx] + tanhf(v);
            }
        }
    }
}

// ---------------------------------------------------------------------------
// Flash attention (bf16 in/out, f32 online softmax). D_K = 64, H = 16.
// grid = (B*H, T/128); 8 waves/block; each wave owns 16 query rows.
// Streams 32-key chunks: S = Q(16x64)K^T, online softmax, O += P(16x32)V.
// ---------------------------------------------------------------------------
template <int MASK2D>
__global__ __launch_bounds__(256)
void flash_attn_kernel(const bf16* __restrict__ Qm, const bf16* __restrict__ Km,
                       const bf16* __restrict__ Vm, const int* __restrict__ mask,
                       bf16* __restrict__ Om, int Tq, int Skv) {
    __shared__ alignas(16) bf16 ldsK[32 * 64];       // [key][feat] row-major
    __shared__ alignas(16) bf16 ldsVt[64 * 32];      // [feat][key] transposed
    __shared__ alignas(16) bf16 ptile[8][16 * 32];   // per-wave P tile

    const int tid = threadIdx.x, lane = tid & 31, wid = tid >> 5;
    const int head = blockIdx.x & 15, b = blockIdx.x >> 4;
    const int qbase = blockIdx.y * 128 + wid * 16;   // query row within batch
    const size_t qrow0 = (size_t)b * Tq + qbase;

    // Q fragments (A-layout, straight from global, 64 features = 2 K-slices)
    v16bf qf[2];
    qf[0] = load_a_frag(Qm + qrow0 * 1024 + head * 64, 0, 1024, lane);
    qf[1] = load_a_frag(Qm + qrow0 * 1024 + head * 64 + 32, 0, 1024, lane);

    float m[8], l[8];
    v8f o[4];
#pragma unroll
    for (int r = 0; r < 8; ++r) { m[r] = -1e30f; l[r] = 0.0f; }
#pragma unroll
    for (int t = 0; t < 4; ++t) o[t] = zero8();

    const float scale = 0.125f;   // 1/sqrt(64)
    const int rowoff = (lane >> 4) * 8;
    const int cl = lane & 15;

    // staging coordinates (loop-invariant)
    const int skey = tid >> 3;                 // K-tile: key row
    const int sf0  = (tid & 7) * 8;            // K-tile: feat
    const int vkey = (tid >> 4) * 2;           // V-tile: key pair
    const int vf0  = (tid & 15) * 4;           // V-tile: 4 feats

    for (int j = 0; j < Skv; j += 32) {
        __syncthreads();
        {   // stage K row-major (one b128 per thread, DMA when available)
            const size_t krow =
                ((size_t)b * Skv + j + skey) * 1024 + head * 64 + sf0;
#if HAS_ASYNC_LDS
            async_copy_b128(Km + krow, ldsK + skey * 64 + sf0);
#else
            *(uint4*)(ldsK + skey * 64 + sf0) = *(const uint4*)(Km + krow);
#endif
        }
        {   // stage V transposed: pack key pairs into b32 stores
            const size_t v0 = ((size_t)b * Skv + j + vkey) * 1024 + head * 64 + vf0;
            const uint2 a0 = *(const uint2*)(Vm + v0);
            const uint2 a1 = *(const uint2*)(Vm + v0 + 1024);
            const unsigned short* s0 = (const unsigned short*)&a0;
            const unsigned short* s1 = (const unsigned short*)&a1;
#pragma unroll
            for (int i = 0; i < 4; ++i) {
                const unsigned pk = (unsigned)s0[i] | ((unsigned)s1[i] << 16);
                *(unsigned*)(ldsVt + (vf0 + i) * 32 + vkey) = pk;
            }
        }
        if (j + 32 < Skv) {
            __builtin_prefetch(
                Km + ((size_t)b * Skv + j + 32 + skey) * 1024 + head * 64, 0, 1);
            __builtin_prefetch(
                Vm + ((size_t)b * Skv + j + 32 + skey) * 1024 + head * 64, 0, 1);
        }
#if HAS_ASYNC_LDS
        async_wait0();
#endif
        __syncthreads();

        // S tile 16x32 as two 16x16 WMMA accumulations (K split 32+32)
        v8f s[2];
#pragma unroll
        for (int t = 0; t < 2; ++t) {
            v8f c = zero8();
            c = wmma_bf16(qf[0], load_b_fragT(ldsK, t * 16, 64, lane), c);
            c = wmma_bf16(qf[1], load_b_fragT(ldsK + 32, t * 16, 64, lane), c);
            s[t] = c;
        }

        // online softmax; row stats reduce across 16-lane halves
#pragma unroll
        for (int r = 0; r < 8; ++r) {
            const int qr = qbase + rowoff + r;
            float s0 = s[0][r] * scale;
            float s1 = s[1][r] * scale;
            const int k0i = j + cl, k1i = j + 16 + cl;
            if constexpr (MASK2D) {
                const size_t mb = (size_t)b * Tq * Skv + (size_t)qr * Skv;
                if (mask[mb + k0i] == 0) s0 = -1e9f;
                if (mask[mb + k1i] == 0) s1 = -1e9f;
            } else {
                const size_t mb = (size_t)b * Skv;
                if (mask[mb + k0i] == 0) s0 = -1e9f;
                if (mask[mb + k1i] == 0) s1 = -1e9f;
            }
            float rmax = fmaxf(s0, s1);
            for (int d = 1; d < 16; d <<= 1)
                rmax = fmaxf(rmax, __shfl_xor(rmax, d, 32));
            const float newm = fmaxf(m[r], rmax);
            const float alpha = expf(m[r] - newm);
            const float p0 = expf(s0 - newm);
            const float p1 = expf(s1 - newm);
            float rsum = p0 + p1;
            for (int d = 1; d < 16; d <<= 1) rsum += __shfl_xor(rsum, d, 32);
            l[r] = l[r] * alpha + rsum;
            m[r] = newm;
            o[0][r] *= alpha; o[1][r] *= alpha;
            o[2][r] *= alpha; o[3][r] *= alpha;
            const int prow = rowoff + r;
            ptile[wid][prow * 32 + cl]      = (bf16)p0;
            ptile[wid][prow * 32 + cl + 16] = (bf16)p1;
        }

        // O(16x64) += P(16x32) x V(32x64)
        const v16bf pa = load_a_frag(ptile[wid], 0, 32, lane);
#pragma unroll
        for (int t2 = 0; t2 < 4; ++t2)
            o[t2] = wmma_bf16(pa, load_b_fragT(ldsVt, t2 * 16, 32, lane), o[t2]);
    }

    // normalize and write bf16 context
#pragma unroll
    for (int t2 = 0; t2 < 4; ++t2) {
#pragma unroll
        for (int r = 0; r < 8; ++r) {
            const float inv = 1.0f / l[r];
            const size_t idx = ((size_t)b * Tq + qbase + rowoff + r) * 1024 +
                               head * 64 + t2 * 16 + cl;
            Om[idx] = (bf16)(o[t2][r] * inv);
        }
    }
}

// ---------------------------------------------------------------------------
// Host-side orchestration
// ---------------------------------------------------------------------------
extern "C" void kernel_launch(void* const* d_in, const int* in_sizes, int n_in,
                              void* d_out, int out_size, void* d_ws, size_t ws_size,
                              hipStream_t stream) {
    (void)in_sizes; (void)n_in; (void)out_size; (void)ws_size;

    constexpr int D = 1024, B = 4, T = 1024, S = 1024, H = 16;
    constexpr int M = B * T;                 // 4096 rows
    constexpr size_t WELEM = (size_t)D * D;  // 1M
    constexpr size_t AELEM = (size_t)M * D;  // 4M

    const float* z        = (const float*)d_in[0];
    const float* eo       = (const float*)d_in[1];
    const int*   src_mask = (const int*)d_in[2];
    const int*   trg_mask = (const int*)d_in[3];
    const float* n1a = (const float*)d_in[4];  const float* n1b = (const float*)d_in[5];
    const float* n2a = (const float*)d_in[6];  const float* n2b = (const float*)d_in[7];
    const float* n3a = (const float*)d_in[8];  const float* n3b = (const float*)d_in[9];
    // attn1: q(10,11) k(12,13) v(14,15) out(16,17); attn2: 18..25
    // ff: linear_1(26,27) mean(28,29) [log_std(30,31) unused in output]
    const float* wq1f = (const float*)d_in[10]; const float* bq1 = (const float*)d_in[11];
    const float* wk1f = (const float*)d_in[12]; const float* bk1 = (const float*)d_in[13];
    const float* wv1f = (const float*)d_in[14]; const float* bv1 = (const float*)d_in[15];
    const float* wo1f = (const float*)d_in[16]; const float* bo1 = (const float*)d_in[17];
    const float* wq2f = (const float*)d_in[18]; const float* bq2 = (const float*)d_in[19];
    const float* wk2f = (const float*)d_in[20]; const float* bk2 = (const float*)d_in[21];
    const float* wv2f = (const float*)d_in[22]; const float* bv2 = (const float*)d_in[23];
    const float* wo2f = (const float*)d_in[24]; const float* bo2 = (const float*)d_in[25];
    const float* wl1f = (const float*)d_in[26]; const float* bl1 = (const float*)d_in[27];
    const float* wmuf = (const float*)d_in[28]; const float* bmu = (const float*)d_in[29];

    // workspace layout
    char* ws = (char*)d_ws;
    size_t off = 0;
    auto alloc = [&](size_t bytes) -> void* {
        void* p = ws + off;
        off += (bytes + 255) & ~(size_t)255;
        return p;
    };
    bf16* wq1 = (bf16*)alloc(WELEM * 2);  bf16* wk1 = (bf16*)alloc(WELEM * 2);
    bf16* wv1 = (bf16*)alloc(WELEM * 2);  bf16* wo1 = (bf16*)alloc(WELEM * 2);
    bf16* wq2 = (bf16*)alloc(WELEM * 2);  bf16* wk2 = (bf16*)alloc(WELEM * 2);
    bf16* wv2 = (bf16*)alloc(WELEM * 2);  bf16* wo2 = (bf16*)alloc(WELEM * 2);
    bf16* wl1 = (bf16*)alloc(WELEM * 2);  bf16* wmu = (bf16*)alloc(WELEM * 2);
    bf16* x2b   = (bf16*)alloc(AELEM * 2);
    bf16* qb    = (bf16*)alloc(AELEM * 2);
    bf16* kb    = (bf16*)alloc(AELEM * 2);
    bf16* vb    = (bf16*)alloc(AELEM * 2);
    bf16* attnb = (bf16*)alloc(AELEM * 2);
    bf16* eb    = (bf16*)alloc(AELEM * 2);
    float* x1   = (float*)alloc(AELEM * 4);

    // 1) convert weights + e_outputs to bf16
    const float* wsrc[10] = {wq1f, wk1f, wv1f, wo1f, wq2f, wk2f, wv2f, wo2f, wl1f, wmuf};
    bf16* wdst[10]        = {wq1,  wk1,  wv1,  wo1,  wq2,  wk2,  wv2,  wo2,  wl1,  wmu};
    for (int i = 0; i < 10; ++i)
        f32_to_bf16_kernel<<<512, 256, 0, stream>>>(wsrc[i], wdst[i], (int)WELEM);
    f32_to_bf16_kernel<<<1024, 256, 0, stream>>>(eo, eb, (int)AELEM);

    const dim3 gemm_grid(M / 128, D / 128);   // 32 x 8
    const dim3 attn_grid(B * H, T / 128);     // 64 x 8

    // 2) x2 = norm1(z)
    layernorm_kernel<<<M, 256, 0, stream>>>(z, n1a, n1b, x2b);
    // 3) self-attn projections
    gemm_bf16_kernel<0><<<gemm_grid, 256, 0, stream>>>(x2b, wq1, bq1, nullptr, nullptr, qb, M, D, D);
    gemm_bf16_kernel<0><<<gemm_grid, 256, 0, stream>>>(x2b, wk1, bk1, nullptr, nullptr, kb, M, D, D);
    gemm_bf16_kernel<0><<<gemm_grid, 256, 0, stream>>>(x2b, wv1, bv1, nullptr, nullptr, vb, M, D, D);
    // 4) self-attention (trg_mask is [B,T,T])
    flash_attn_kernel<1><<<attn_grid, 256, 0, stream>>>(qb, kb, vb, trg_mask, attnb, T, T);
    // 5) x1 = z + out_proj(attn)
    gemm_bf16_kernel<1><<<gemm_grid, 256, 0, stream>>>(attnb, wo1, bo1, z, x1, nullptr, M, D, D);
    // 6) x2 = norm2(x1)
    layernorm_kernel<<<M, 256, 0, stream>>>(x1, n2a, n2b, x2b);
    // 7) cross-attn projections (K/V from e_outputs)
    gemm_bf16_kernel<0><<<gemm_grid, 256, 0, stream>>>(x2b, wq2, bq2, nullptr, nullptr, qb, M, D, D);
    gemm_bf16_kernel<0><<<gemm_grid, 256, 0, stream>>>(eb,  wk2, bk2, nullptr, nullptr, kb, M, D, D);
    gemm_bf16_kernel<0><<<gemm_grid, 256, 0, stream>>>(eb,  wv2, bv2, nullptr, nullptr, vb, M, D, D);
    // 8) cross-attention (src_mask is [B,1,S], broadcast over queries)
    flash_attn_kernel<0><<<attn_grid, 256, 0, stream>>>(qb, kb, vb, src_mask, attnb, T, S);
    // 9) x1 = x1 + out_proj(attn)   (in-place residual: same-index RMW)
    gemm_bf16_kernel<1><<<gemm_grid, 256, 0, stream>>>(attnb, wo2, bo2, x1, x1, nullptr, M, D, D);
    // 10) x2 = norm3(x1)
    layernorm_kernel<<<M, 256, 0, stream>>>(x1, n3a, n3b, x2b);
    // 11) h = relu(linear_1(x2))  (reuse qb)
    gemm_bf16_kernel<2><<<gemm_grid, 256, 0, stream>>>(x2b, wl1, bl1, nullptr, nullptr, qb, M, D, D);
    // 12) out = x1 + tanh(mean(h))
    gemm_bf16_kernel<3><<<gemm_grid, 256, 0, stream>>>(qb, wmu, bmu, x1, (float*)d_out, nullptr, M, D, D);
}
